// NaiveLSTM_42305427865876
// MI455X (gfx1250) — matrix operand order repaired and checked by reading, not compiled, not run on hardware
//
#include <hip/hip_runtime.h>
#include <cstddef>
#include <cstdint>

// LSTM forward for MI455X (gfx1250, wave32, WMMA).
//   (1) pack weights -> bf16 TRANSPOSED ([N][K]) so both WMMA operands are
//       K-contiguous per lane -> fragment loads are 2x ds_load_b128 each
//   (2) xg = x @ Wg + bg        : bf16 WMMA GEMM, f32 accum -> ws
//   (3) 512x lstm_step kernels  : g = xg[t] + h @ Ug (bf16 WMMA), gates in f32
//   (4) y  = hs @ Wy + by       : bf16 WMMA GEMM -> d_out
// Pipelining: fetch next tile (async global->LDS if available, else into
// registers), compute current tile from LDS, commit, ONE barrier per K-step.

typedef __attribute__((ext_vector_type(16))) __bf16 v16bf;
typedef __attribute__((ext_vector_type(8)))  __bf16 v8bf;
typedef __attribute__((ext_vector_type(4)))  __bf16 v4bf;
typedef __attribute__((ext_vector_type(8)))  float  v8f;
typedef __attribute__((ext_vector_type(4)))  int    v4i;

#define NB   64      // batch
#define NT   512     // timesteps
#define ND   1024    // input dim
#define NH   1024    // hidden
#define NG   4096    // 4*H
#define NO   1024    // output

// -------- CDNA5 async global->LDS (ASYNCcnt) path, guarded ------------------
#if defined(__HIP_DEVICE_COMPILE__) && \
    __has_builtin(__builtin_amdgcn_global_load_async_to_lds_b128) && \
    __has_builtin(__builtin_amdgcn_s_wait_asynccnt)
#define ASYNC_LDS 1
#else
#define ASYNC_LDS 0
#endif

#if ASYNC_LDS
__device__ inline void async_copy16(void* lds, const void* g) {
  __builtin_amdgcn_global_load_async_to_lds_b128(
      (__attribute__((address_space(1))) v4i*)(uintptr_t)g,
      (__attribute__((address_space(3))) v4i*)(unsigned)(uintptr_t)lds,
      0, 0);
}
#endif

__device__ inline void wait_async_copies() {
#if ASYNC_LDS
  __builtin_amdgcn_s_wait_asynccnt(0);
#endif
}

// ---------------- WMMA fragment helpers (wave32 layouts per CDNA5 ISA) ------

// A-matrix 16x32 bf16 frag from LDS tile (row-major, `stride` bf16 per row;
// stride*2 must be a multiple of 16B). lane L: m = m0 + L%16;
// elems 0..7 = K[8*half .. +7], elems 8..15 = K[16+8*half .. +7].
__device__ inline v16bf load_frag_a(const __bf16* p, int m0, int stride) {
  const int lid  = threadIdx.x & 31;
  const int half = lid >> 4;
  const __bf16* row = p + (size_t)(m0 + (lid & 15)) * stride + half * 8;
  v8bf lo = *(const v8bf*)(row);
  v8bf hi = *(const v8bf*)(row + 16);
  return __builtin_shufflevector(lo, hi, 0,1,2,3,4,5,6,7,8,9,10,11,12,13,14,15);
}

// B-matrix 32x16 bf16 frag from a TRANSPOSED LDS tile laid out [n][k]
// (`stride` bf16 per n-row). lane L: n = n0 + L%16;
// elems 0..7 = K[8*(L/16) .. +7], elems 8..15 = K[16+8*(L/16) .. +7]  ->  2x b128.
__device__ inline v16bf load_frag_bt(const __bf16* p, int n0, int stride) {
  const int lid = threadIdx.x & 31;
  const __bf16* row = p + (size_t)(n0 + (lid & 15)) * stride + (lid >> 4) * 8;
  v8bf lo = *(const v8bf*)(row);
  v8bf hi = *(const v8bf*)(row + 16);
  return __builtin_shufflevector(lo, hi, 0,1,2,3,4,5,6,7,8,9,10,11,12,13,14,15);
}

__device__ inline float fast_sigmoid(float x) { return 1.0f / (1.0f + __expf(-x)); }

// ---------------- prep kernels ---------------------------------------------

// Transposed gate pack: outT[(g*NH + h)*1024 + k] = W_gate_g[k*NH + h],
// gate order j,i,f,o. K is always 1024 here (D or H).
__global__ void k_pack_gates_t(const float* __restrict__ Wj, const float* __restrict__ Wi,
                               const float* __restrict__ Wf, const float* __restrict__ Wo,
                               __bf16* __restrict__ outT) {
  int idx = blockIdx.x * blockDim.x + threadIdx.x;   // over NG*1024
  if (idx >= NG * 1024) return;
  int n = idx >> 10, k = idx & 1023;
  int g = n >> 10, h = n & 1023;
  const float* src = (g == 0) ? Wj : (g == 1) ? Wi : (g == 2) ? Wf : Wo;
  outT[idx] = (__bf16)src[(size_t)k * NH + h];
}

// outT[o*NH + h] = W_y[h*NO + o]
__global__ void k_pack_wy_t(const float* __restrict__ Wy, __bf16* __restrict__ outT) {
  int idx = blockIdx.x * blockDim.x + threadIdx.x;
  if (idx >= NO * NH) return;
  int o = idx >> 10, h = idx & 1023;
  outT[idx] = (__bf16)Wy[(size_t)h * NO + o];
}

__global__ void k_bias(const float* __restrict__ bij, const float* __restrict__ bhj,
                       const float* __restrict__ bii, const float* __restrict__ bhi,
                       const float* __restrict__ bif_, const float* __restrict__ bhf,
                       const float* __restrict__ bio, const float* __restrict__ bho,
                       float* __restrict__ bg) {
  int i = blockIdx.x * blockDim.x + threadIdx.x;
  if (i < NH) {
    bg[i]          = bij[i] + bhj[i];
    bg[NH + i]     = bii[i] + bhi[i];
    bg[2 * NH + i] = bif_[i] + bhf[i];
    bg[3 * NH + i] = bio[i] + bho[i];
  }
}

__global__ void k_init(const float* __restrict__ h0, const float* __restrict__ c0,
                       float* __restrict__ hT, float* __restrict__ cT,
                       __bf16* __restrict__ hbf, int n) {
  int i = blockIdx.x * blockDim.x + threadIdx.x;
  if (i < n) { hT[i] = h0[i]; cT[i] = c0[i]; hbf[i] = (__bf16)h0[i]; }
}

// ---------------- batched GEMM (xg pre-pass & y post-pass) -----------------
// C[M,N] = A[M,K] * BT[N,K]^T + bias[N]; A fp32 (cast on the fly) or bf16.
// Tile 128x128, 256 threads = 8 waves (2x4), wave tile 64x32 (4x2 WMMA accs).

#define TM 128
#define TN 128
#define TK 32
#define AST 40    // LDS A row stride (bf16): 80B, 16B aligned, conflict-free
#define BTS 40    // LDS B^T row stride (bf16): 80B

template <bool AF32>
__global__ __launch_bounds__(256) void k_gemm(const void* __restrict__ Aptr,
                                              const __bf16* __restrict__ BT,
                                              const float* __restrict__ bias,
                                              float* __restrict__ Cout,
                                              int N, int K) {
  __shared__ __align__(16) __bf16 ldsA[2][TM * AST];
  __shared__ __align__(16) __bf16 ldsBT[2][TN * BTS];
  const int tid = threadIdx.x;
  const int lid = tid & 31, wid = tid >> 5;
  const int wm = wid >> 2;            // 0..1 : 64 rows per wave
  const int wn = wid & 3;             // 0..3 : 32 cols per wave
  const int row0 = blockIdx.y * TM;
  const int col0 = blockIdx.x * TN;

  float4 ra[4];   // staging regs (A: 4 chunks f32 / 2 chunks bf16)
  float4 rb[2];   // staging regs (B^T: 2 chunks) -- unused on async path

  // Issue next tile's global traffic. Async path copies straight to LDS.
  auto fetch = [&](int buf, int kk) {
    (void)buf;
    if (AF32) {
      const float* A = (const float*)Aptr;
#pragma unroll
      for (int u = 0; u < 4; ++u) {
        int idx = (tid + u * 256) * 4, r = idx >> 5, c = idx & 31;
        ra[u] = *(const float4*)(A + (size_t)(row0 + r) * K + kk + c);
      }
    } else {
      const __bf16* A = (const __bf16*)Aptr;
#pragma unroll
      for (int u = 0; u < 2; ++u) {
        int idx = (tid + u * 256) * 8, r = idx >> 5, c = idx & 31;
#if ASYNC_LDS
        async_copy16(&ldsA[buf][r * AST + c], A + (size_t)(row0 + r) * K + kk + c);
#else
        ra[u] = *(const float4*)(A + (size_t)(row0 + r) * K + kk + c);
#endif
      }
    }
#pragma unroll
    for (int u = 0; u < 2; ++u) {
      int idx = (tid + u * 256) * 8, n = idx >> 5, k = idx & 31;
#if ASYNC_LDS
      async_copy16(&ldsBT[buf][n * BTS + k], BT + (size_t)(col0 + n) * K + kk + k);
#else
      rb[u] = *(const float4*)(BT + (size_t)(col0 + n) * K + kk + k);
#endif
    }
  };

  // Commit register-staged chunks to LDS (after compute, so the loadcnt wait
  // lands behind the WMMA block). Empty for fully-async tiles.
  auto commit = [&](int buf) {
    (void)buf;
    if (AF32) {
#pragma unroll
      for (int u = 0; u < 4; ++u) {
        int idx = (tid + u * 256) * 4, r = idx >> 5, c = idx & 31;
        v4bf pk;
        pk[0] = (__bf16)ra[u].x; pk[1] = (__bf16)ra[u].y;
        pk[2] = (__bf16)ra[u].z; pk[3] = (__bf16)ra[u].w;
        *(v4bf*)&ldsA[buf][r * AST + c] = pk;
      }
    }
#if !ASYNC_LDS
    else {
#pragma unroll
      for (int u = 0; u < 2; ++u) {
        int idx = (tid + u * 256) * 8, r = idx >> 5, c = idx & 31;
        *(float4*)&ldsA[buf][r * AST + c] = ra[u];
      }
    }
#pragma unroll
    for (int u = 0; u < 2; ++u) {
      int idx = (tid + u * 256) * 8, n = idx >> 5, k = idx & 31;
      *(float4*)&ldsBT[buf][n * BTS + k] = rb[u];
    }
#endif
  };

  v8f acc[4][2] = {};

  fetch(0, 0);
  commit(0);
  wait_async_copies();
  __syncthreads();
  for (int kk = 0; kk < K; kk += TK) {
    const int cur = (kk / TK) & 1;
    const bool more = (kk + TK) < K;
    if (more) fetch(cur ^ 1, kk + TK);

    v16bf af[4], bfr[2];
#pragma unroll
    for (int i = 0; i < 4; ++i) af[i] = load_frag_a(ldsA[cur], wm * 64 + i * 16, AST);
#pragma unroll
    for (int j = 0; j < 2; ++j) bfr[j] = load_frag_bt(ldsBT[cur], wn * 32 + j * 16, BTS);
#pragma unroll
    for (int i = 0; i < 4; ++i)
#pragma unroll
      for (int j = 0; j < 2; ++j)
        acc[i][j] = __builtin_amdgcn_wmma_f32_16x16x32_bf16(
            false, af[i], false, bfr[j], (short)0, acc[i][j], false, false);

    if (more) commit(cur ^ 1);
    wait_async_copies();
    __syncthreads();
  }

  // epilogue: + bias, store f32 (C/D layout: VGPR r -> m = r + 8*half, lane -> n)
  const int half = lid >> 4, nl = lid & 15;
#pragma unroll
  for (int i = 0; i < 4; ++i)
#pragma unroll
    for (int j = 0; j < 2; ++j) {
      int n = col0 + wn * 32 + j * 16 + nl;
      float bb = bias[n];
#pragma unroll
      for (int r = 0; r < 8; ++r) {
        int m = row0 + wm * 64 + i * 16 + half * 8 + r;
        Cout[(size_t)m * N + n] = acc[i][j][r] + bb;
      }
    }
}

// ---------------- recurrent step -------------------------------------------
// Each workgroup (8 waves): same 32 hidden columns across ALL 4 gates, so the
// LSTM pointwise math is lane-local in the C-fragment layout.

__global__ __launch_bounds__(256) void k_lstm_step(
    const __bf16* __restrict__ h_in, const __bf16* __restrict__ UgT,
    const float* __restrict__ xg, int t,
    float* __restrict__ h_state, float* __restrict__ c_state,
    __bf16* __restrict__ h_out, __bf16* __restrict__ hs_bf) {
  __shared__ __align__(16) __bf16 ldsH[2][NB * AST];      // 64x32 h tile
  __shared__ __align__(16) __bf16 ldsU[2][4][32 * BTS];   // 4 gates x (32n x 32k)
  const int tid = threadIdx.x;
  const int lid = tid & 31, wid = tid >> 5;
  const int mt = wid >> 1;                   // 0..3 : rows mt*16..+15
  const int nt = wid & 1;                    // 0..1 : cols nt*16..+15
  const int colB = blockIdx.x * 32;          // base hidden column of this WG

  float4 rh, ru0, ru1;

  auto fetch = [&](int buf, int kk) {
    (void)buf;
    { int idx = tid * 8, r = idx >> 5, c = idx & 31;
#if ASYNC_LDS
      async_copy16(&ldsH[buf][r * AST + c], h_in + (size_t)r * NH + kk + c);
#else
      rh = *(const float4*)(h_in + (size_t)r * NH + kk + c);
#endif
    }
#pragma unroll
    for (int u = 0; u < 2; ++u) {
      int idx = (tid + u * 256) * 8, g = idx >> 10, rem = idx & 1023;
      int n = rem >> 5, k = rem & 31;
#if ASYNC_LDS
      async_copy16(&ldsU[buf][g][n * BTS + k],
                   UgT + (size_t)(g * NH + colB + n) * NH + kk + k);
#else
      (u ? ru1 : ru0) = *(const float4*)(UgT + (size_t)(g * NH + colB + n) * NH + kk + k);
#endif
    }
  };

  auto commit = [&](int buf) {
    (void)buf;
#if !ASYNC_LDS
    { int idx = tid * 8, r = idx >> 5, c = idx & 31;
      *(float4*)&ldsH[buf][r * AST + c] = rh; }
#pragma unroll
    for (int u = 0; u < 2; ++u) {
      int idx = (tid + u * 256) * 8, g = idx >> 10, rem = idx & 1023;
      int n = rem >> 5, k = rem & 31;
      *(float4*)&ldsU[buf][g][n * BTS + k] = (u ? ru1 : ru0);
    }
#endif
  };

  v8f acc[4] = {};

  fetch(0, 0);
  commit(0);
  wait_async_copies();
  __syncthreads();
  for (int kk = 0; kk < NH; kk += TK) {
    const int cur = (kk / TK) & 1;
    const bool more = (kk + TK) < NH;
    if (more) fetch(cur ^ 1, kk + TK);

    v16bf hf = load_frag_a(ldsH[cur], mt * 16, AST);
#pragma unroll
    for (int g = 0; g < 4; ++g) {
      v16bf uf = load_frag_bt(ldsU[cur][g], nt * 16, BTS);
      acc[g] = __builtin_amdgcn_wmma_f32_16x16x32_bf16(
          false, hf, false, uf, (short)0, acc[g], false, false);
    }

    if (more) commit(cur ^ 1);
    wait_async_copies();
    __syncthreads();
  }

  // pointwise LSTM epilogue (lane-local in C-fragment layout)
  const int half = lid >> 4, nl = lid & 15;
  const int cH = colB + nt * 16 + nl;        // hidden column 0..1023
#pragma unroll
  for (int r = 0; r < 8; ++r) {
    int b = mt * 16 + half * 8 + r;          // batch row 0..63
    size_t xrow = ((size_t)b * NT + t) * NG;
    float gj = acc[0][r] + xg[xrow + 0 * NH + cH];
    float gi = acc[1][r] + xg[xrow + 1 * NH + cH];
    float gf = acc[2][r] + xg[xrow + 2 * NH + cH];
    float go = acc[3][r] + xg[xrow + 3 * NH + cH];
    float jj = tanhf(gj);
    float ii = fast_sigmoid(gi);
    float ff = fast_sigmoid(gf);
    float oo = fast_sigmoid(go);
    size_t sidx = (size_t)b * NH + cH;
    float cn = ff * c_state[sidx] + ii * jj;
    float hn = oo * tanhf(cn);
    c_state[sidx] = cn;
    h_state[sidx] = hn;
    __bf16 hb = (__bf16)hn;
    h_out[sidx] = hb;
    hs_bf[((size_t)b * NT + t) * NH + cH] = hb;
  }
}

// ---------------- host ------------------------------------------------------

extern "C" void kernel_launch(void* const* d_in, const int* in_sizes, int n_in,
                              void* d_out, int out_size, void* d_ws, size_t ws_size,
                              hipStream_t stream) {
  (void)in_sizes; (void)n_in; (void)out_size; (void)ws_size;
  const float* x    = (const float*)d_in[0];
  const float* h0   = (const float*)d_in[1];
  const float* c0   = (const float*)d_in[2];
  const float* W_i  = (const float*)d_in[3];
  const float* W_f  = (const float*)d_in[4];
  const float* W_j  = (const float*)d_in[5];
  const float* W_o  = (const float*)d_in[6];
  const float* U_i  = (const float*)d_in[7];
  const float* U_f  = (const float*)d_in[8];
  const float* U_j  = (const float*)d_in[9];
  const float* U_o  = (const float*)d_in[10];
  const float* b_ii = (const float*)d_in[11];
  const float* b_if = (const float*)d_in[12];
  const float* b_ij = (const float*)d_in[13];
  const float* b_io = (const float*)d_in[14];
  const float* b_hi = (const float*)d_in[15];
  const float* b_hf = (const float*)d_in[16];
  const float* b_hj = (const float*)d_in[17];
  const float* b_ho = (const float*)d_in[18];
  const float* W_y  = (const float*)d_in[19];
  const float* b_y  = (const float*)d_in[20];

  float* out = (float*)d_out;
  float* y   = out;                                   // [B,T,O]
  float* hT  = out + (size_t)NB * NT * NO;            // [B,H]  (live h state)
  float* cT  = hT + (size_t)NB * NH;                  // [B,H]  (live c state)

  char* ws = (char*)d_ws;
  size_t off = 0;
  auto take = [&](size_t bytes) -> char* {
    char* p = ws + off;
    off += (bytes + 255) & ~(size_t)255;
    return p;
  };
  __bf16* wgT   = (__bf16*)take((size_t)NG * ND * 2);          //  8 MB  [4H][D]
  __bf16* ugT   = (__bf16*)take((size_t)NG * NH * 2);          //  8 MB  [4H][H]
  __bf16* wyT   = (__bf16*)take((size_t)NO * NH * 2);          //  2 MB  [O][H]
  float*  bg    = (float*) take((size_t)NG * 4);
  __bf16* hbuf0 = (__bf16*)take((size_t)NB * NH * 2);
  __bf16* hbuf1 = (__bf16*)take((size_t)NB * NH * 2);
  __bf16* hsb   = (__bf16*)take((size_t)NB * NT * NH * 2);     // 64 MB
  float*  xg    = (float*) take((size_t)NB * NT * NG * 4);     // 512 MB

  // (1) transposed weight packing + fused biases + state init
  k_pack_gates_t<<<dim3((NG * 1024 + 255) / 256), 256, 0, stream>>>(W_j, W_i, W_f, W_o, wgT);
  k_pack_gates_t<<<dim3((NG * 1024 + 255) / 256), 256, 0, stream>>>(U_j, U_i, U_f, U_o, ugT);
  k_pack_wy_t   <<<dim3((NO * NH + 255) / 256), 256, 0, stream>>>(W_y, wyT);
  k_bias        <<<dim3(4), 256, 0, stream>>>(b_ij, b_hj, b_ii, b_hi, b_if, b_hf, b_io, b_ho, bg);
  k_init        <<<dim3((NB * NH + 255) / 256), 256, 0, stream>>>(h0, c0, hT, cT, hbuf0, NB * NH);

  // (2) xg = x @ Wg + bg   (M=32768, N=4096, K=1024)
  k_gemm<true><<<dim3(NG / TN, (NB * NT) / TM), 256, 0, stream>>>(x, wgT, bg, xg, NG, ND);

  // (3) recurrence: one kernel per timestep; h bf16 double-buffered
  for (int t = 0; t < NT; ++t) {
    const __bf16* hin = (t & 1) ? hbuf1 : hbuf0;
    __bf16*       hout = (t & 1) ? hbuf0 : hbuf1;
    k_lstm_step<<<dim3(NH / 32), 256, 0, stream>>>(hin, ugT, xg, t, hT, cT, hout, hsb);
  }

  // (4) y = hs @ Wy + by   (M=32768, N=1024, K=1024)
  k_gemm<false><<<dim3(NO / TN, (NB * NT) / TM), 256, 0, stream>>>(hsb, wyT, b_y, y, NO, NH);
}
